// rnn_reg_63144609185915
// MI455X (gfx1250) — matrix operand order repaired and checked by reading, not compiled, not run on hardware
//
#include <hip/hip_runtime.h>

typedef __attribute__((ext_vector_type(16))) __bf16 v16bf;
typedef __attribute__((ext_vector_type(8)))  float  v8f;

#define HD 32
#define WAVES_PER_BLOCK 4
// ds_swizzle group-of-32: offset15=0, or=0x10, xor=0, and=0x1f -> lane l reads lane l|16 (head reduce only)
#define SWAPX16 0x021f

// single-instruction packed f32->bf16 convert
static __device__ __forceinline__ unsigned cvt_pk_bf16(float a, float b) {
    unsigned r;
    asm("v_cvt_pk_bf16_f32 %0, %1, %2" : "=v"(r) : "v"(a), "v"(b));
    return r;
}
// packed bf16 relu: one op for two elements
static __device__ __forceinline__ unsigned pk_relu_bf16(unsigned a) {
    unsigned r;
    asm("v_pk_max_num_bf16 %0, %1, 0" : "=v"(r) : "v"(a));
    return r;
}
// K-group swap helper (not called: A's K columns are unpermuted):
// maps K groups [8,16) and [16,24) onto each other, others unchanged.
static __device__ __forceinline__ int kswap(int k) {
    const int g = (k >> 3) & 3;
    return g == 1 ? k + 8 : (g == 2 ? k - 8 : k);
}

__global__ __launch_bounds__(128) void rnn_relu_wmma_kernel(
    const float* __restrict__ x, const float* __restrict__ h_init,
    const float* __restrict__ W_ih, const float* __restrict__ W_hh,
    const float* __restrict__ b_ih, const float* __restrict__ b_hh,
    const float* __restrict__ W_reg, const float* __restrict__ b_reg,
    float* __restrict__ out, int Bn, int T)
{
    const int lane = threadIdx.x & 31;
    const int half = lane >> 4;     // 0: lanes 0-15, 1: lanes 16-31
    const int c    = lane & 15;     // batch column within the 16-row tile

    const int tile = blockIdx.x * WAVES_PER_BLOCK + (threadIdx.x >> 5);
    const int b0   = tile * 16;
    if (b0 >= Bn) return;           // whole wave exits together: EXEC stays all-1s

    // ---- A operands (loop-invariant): W_hh rows, K columns unpermuted ----
    // element e -> physical K slot: (e<8?0:16) + 8*half + (e&7)
    v16bf A0, A1;
    #pragma unroll
    for (int e = 0; e < 16; ++e) {
        const int kp = ((e < 8) ? 0 : 16) + half * 8 + (e & 7);
        const int ku = kp;
        A0[e] = (__bf16)W_hh[c * HD + ku];          // A row m = lane&15
        A1[e] = (__bf16)W_hh[(c + 16) * HD + ku];
    }

    // ---- transposed input-projection constants: rows hr = v + 8*half (+16) ----
    float wih0[8], wih1[8], bb0[8], bb1[8];
    #pragma unroll
    for (int v = 0; v < 8; ++v) {
        const int hr = v + 8 * half;
        wih0[v] = W_ih[hr];        bb0[v] = b_ih[hr] + b_hh[hr];
        wih1[v] = W_ih[hr + 16];   bb1[v] = b_ih[hr + 16] + b_hh[hr + 16];
    }

    // ---- h^T carried directly as the bf16 B operand (zero movement per step) ----
    // u[j]   = T0 rows 2j,2j+1 (lo) / 2j+8,2j+9 (hi)   -> B K-slots 2j,2j+1 / 8+2j,9+2j
    // u[j+4] = T1 rows (same pattern, units +16)        -> B K-slots 16+2j.. / 24+2j..
    union { unsigned u[8]; v16bf v; } hb;
    {
        const size_t base = (size_t)(b0 + c) * HD;
        #pragma unroll
        for (int j = 0; j < 4; ++j) {
            const int hr = 2 * j + 8 * half;
            hb.u[j]     = cvt_pk_bf16(h_init[base + hr],      h_init[base + hr + 1]);
            hb.u[j + 4] = cvt_pk_bf16(h_init[base + hr + 16], h_init[base + hr + 17]);
        }
    }

    v8f T0 = {}, T1 = {};                           // f32 D tiles (raw, pre-relu)
    const float* xrow = x + (size_t)(b0 + c) * T;   // one x row per lane (halves duplicate)

    for (int t0 = 0; t0 < T; t0 += 4) {
        const float4 xq = *(const float4*)(xrow + t0);
        const float xs[4] = { xq.x, xq.y, xq.z, xq.w };
        #pragma unroll
        for (int dt = 0; dt < 4; ++dt) {
            // ---- C = xin^T tile: one x scalar per lane, 16 fma (pk_fma'd) ----
            const float xv = xs[dt];
            v8f c0, c1;
            #pragma unroll
            for (int v = 0; v < 8; ++v) {
                c0[v] = fmaf(xv, wih0[v], bb0[v]);
                c1[v] = fmaf(xv, wih1[v], bb1[v]);
            }
            // ---- h_new^T (pre-relu) = W_hh * h^T + xin^T : 2 WMMAs, invariant A ----
            T0 = __builtin_amdgcn_wmma_f32_16x16x32_bf16(false, A0, false, hb.v, (short)0, c0, false, false);
            T1 = __builtin_amdgcn_wmma_f32_16x16x32_bf16(false, A1, false, hb.v, (short)0, c1, false, false);
            // ---- pack + relu straight into next step's B operand ----
            #pragma unroll
            for (int j = 0; j < 4; ++j) {
                hb.u[j]     = pk_relu_bf16(cvt_pk_bf16(T0[2 * j], T0[2 * j + 1]));
                hb.u[j + 4] = pk_relu_bf16(cvt_pk_bf16(T1[2 * j], T1[2 * j + 1]));
            }
        }
    }

    // ---- regression head: out[b0+c] = sum_k relu(h)[c][k]*W_reg[k] + b_reg ----
    float part = 0.0f;
    #pragma unroll
    for (int v = 0; v < 8; ++v) {
        const int k = v + 8 * half;           // T0 holds unit k, T1 holds k+16
        part = fmaf(fmaxf(T0[v], 0.0f), W_reg[k],      part);
        part = fmaf(fmaxf(T1[v], 0.0f), W_reg[k + 16], part);
    }
    union { float f; int i; } pu, sw;
    pu.f = part;
    sw.i = __builtin_amdgcn_ds_swizzle(pu.i, SWAPX16);  // fold in the other half-wave
    if (lane < 16) out[b0 + c] = part + sw.f + b_reg[0];
}

extern "C" void kernel_launch(void* const* d_in, const int* in_sizes, int n_in,
                              void* d_out, int out_size, void* d_ws, size_t ws_size,
                              hipStream_t stream) {
    const float* x      = (const float*)d_in[0];
    const float* h_init = (const float*)d_in[1];
    const float* W_ih   = (const float*)d_in[2];
    const float* W_hh   = (const float*)d_in[3];
    const float* b_ih   = (const float*)d_in[4];
    const float* b_hh   = (const float*)d_in[5];
    const float* W_reg  = (const float*)d_in[6];
    const float* b_reg  = (const float*)d_in[7];

    const int Bn = in_sizes[1] / HD;       // h_init is [1,B,H]
    const int T  = in_sizes[0] / Bn;       // x is [B,T,1]

    const int tiles  = (Bn + 15) / 16;     // one wave per 16 batch rows
    const int blocks = (tiles + WAVES_PER_BLOCK - 1) / WAVES_PER_BLOCK;

    rnn_relu_wmma_kernel<<<blocks, WAVES_PER_BLOCK * 32, 0, stream>>>(
        x, h_init, W_ih, W_hh, b_ih, b_hh, W_reg, b_reg,
        (float*)d_out, Bn, T);
}
